// LongRangeTemporalAttention_21775484191192
// MI455X (gfx1250) — matrix operand order repaired and checked by reading, not compiled, not run on hardware
//
#include <hip/hip_runtime.h>
#include <stdint.h>

// ---------- CDNA5 WMMA types ----------
typedef __bf16 bf16;
typedef __attribute__((ext_vector_type(16))) bf16  v16bf;
typedef __attribute__((ext_vector_type(8)))  bf16  v8bf;
typedef __attribute__((ext_vector_type(8)))  float v8f;

#define WMMA_BF16(a, b, c) \
  __builtin_amdgcn_wmma_f32_16x16x32_bf16(false, (a), false, (b), (short)0, (c), false, false)

// Async global->LDS copies (CDNA5 ASYNCcnt path), guarded for toolchain support.
#if defined(__gfx1250__) && __has_builtin(__builtin_amdgcn_global_load_async_to_lds_b128) && \
    __has_builtin(__builtin_amdgcn_s_wait_asynccnt)
#define USE_ASYNC_LDS 1
#else
#define USE_ASYNC_LDS 0
#endif

#if USE_ASYNC_LDS
// Builtin prototype (from compiler diagnostic): b128 variant moves one
// "int __attribute__((vector_size(16)))" per lane: (AS1 v4i* src, AS3 v4i* dst, imm, cpol)
typedef int v4i __attribute__((vector_size(16)));
typedef __attribute__((address_space(1))) v4i gas_v4i;
typedef __attribute__((address_space(3))) v4i lds_v4i;
__device__ __forceinline__ void async_copy_b128(const void* gsrc, void* ldst) {
  // LDS generic addresses carry the LDS offset in the low 32 bits.
  __builtin_amdgcn_global_load_async_to_lds_b128(
      (gas_v4i*)(uintptr_t)gsrc, (lds_v4i*)(uint32_t)(uintptr_t)ldst, 0, 0);
}
#endif

// ---------- helpers ----------
__device__ __forceinline__ uint16_t f32_to_bf16(float x) {
  uint32_t u = __float_as_uint(x);
  u += 0x7FFFu + ((u >> 16) & 1u);     // round-to-nearest-even
  return (uint16_t)(u >> 16);
}
__device__ __forceinline__ bf16 f32_to_bf16h(float x) {
  uint16_t v = f32_to_bf16(x);
  return __builtin_bit_cast(bf16, v);
}

// A-fragment 16x32 (ISA 7.12.2): lane row = lane&15; lanes<16 hold K [0..7],[16..23],
// lanes>=16 hold K [8..15],[24..31].  p points at row start of the 32-wide K window.
__device__ __forceinline__ v16bf load_frag_A(const uint16_t* p, int lane) {
  int c0 = (lane >> 4) * 8;
  union { v16bf v; v8bf h[2]; } u;
  u.h[0] = *(const v8bf*)(p + c0);
  u.h[1] = *(const v8bf*)(p + c0 + 16);
  return u.v;
}
// B-fragment 32x16: lane col = lane&15; lanes<16 hold K [0..7],[16..23],
// lanes>=16 hold K [8..15],[24..31] (same as A).  colPtr points at column's K-window start.
__device__ __forceinline__ v16bf load_frag_B(const uint16_t* colPtr, int lane) {
  int k0 = (lane >> 4) * 8;
  union { v16bf v; v8bf h[2]; } u;
  u.h[0] = *(const v8bf*)(colPtr + k0);
  u.h[1] = *(const v8bf*)(colPtr + k0 + 16);
  return u.v;
}

// ---------- elementwise conversions ----------
__global__ void convert_f32_bf16(const float* __restrict__ in, uint16_t* __restrict__ out, int n) {
  int i = blockIdx.x * blockDim.x + threadIdx.x;
  if (i < n) out[i] = f32_to_bf16(in[i]);
}
__global__ void convert_pad_f32_bf16(const float* __restrict__ in, uint16_t* __restrict__ out,
                                     int n, int total) {
  int i = blockIdx.x * blockDim.x + threadIdx.x;
  if (i < total) out[i] = (i < n) ? f32_to_bf16(in[i]) : (uint16_t)0;
}

// ---------- projection GEMM:  Y[M=4096, N=1024] = X @ W^T + bias ----------
// mode 0: bf16 out, [B,H,S,hd]   (Q, K)
// mode 1: bf16 out, [B,H,hd,S]   (V transposed for PV B-fragments)
// mode 2: f32  out, [M, N]       (final output projection)
__global__ __launch_bounds__(128)
void proj_gemm(const uint16_t* __restrict__ X, const uint16_t* __restrict__ W,
               const float* __restrict__ bias, void* __restrict__ out, int mode) {
  const int K = 1024, N = 1024;
  int lane = threadIdx.x & 31;
  int wave = (blockIdx.x * blockDim.x + threadIdx.x) >> 5;
  int m0 = (wave >> 4) * 16;     // 256 M-tiles
  int n0 = (wave & 15) * 64;     // 16 waves across N, 64 cols each
  int hi = lane >> 4, colLane = lane & 15;

  const uint16_t* aRow = X + (long)(m0 + colLane) * K;
  const uint16_t* wCol[4];
#pragma unroll
  for (int t = 0; t < 4; ++t) wCol[t] = W + (long)(n0 + t * 16 + colLane) * K;

  // software pipeline: fragments for step ks live in (a,b); next step preloads (an,bn).
  // Unrolled x2 so even/odd iterations alternate buffer roles (no register rotation moves).
  v8f acc[4] = {};
  v16bf a = load_frag_A(aRow, lane);
  v16bf b[4];
#pragma unroll
  for (int t = 0; t < 4; ++t) b[t] = load_frag_B(wCol[t], lane);

#pragma unroll 2
  for (int ks = 0; ks < K; ks += 32) {
    int ksn = (ks + 32 < K) ? ks + 32 : ks;   // clamped (redundant reload on last iter)
    v16bf an = load_frag_A(aRow + ksn, lane);
    v16bf bn[4];
#pragma unroll
    for (int t = 0; t < 4; ++t) bn[t] = load_frag_B(wCol[t] + ksn, lane);
#pragma unroll
    for (int t = 0; t < 4; ++t) acc[t] = WMMA_BF16(a, b[t], acc[t]);
    a = an;
#pragma unroll
    for (int t = 0; t < 4; ++t) b[t] = bn[t];
  }

#pragma unroll
  for (int t = 0; t < 4; ++t) {
    int n = n0 + t * 16 + colLane;
    float bn = bias[n];
#pragma unroll
    for (int j = 0; j < 8; ++j) {
      int m = m0 + j + 8 * hi;
      float v = acc[t][j] + bn;
      if (mode == 2) {
        ((float*)out)[(long)m * N + n] = v;
      } else {
        int bb = m >> 10, s = m & 1023;
        int h = n >> 6,  d = n & 63;
        long idx = (mode == 0) ? ((long)((bb * 16 + h) * 1024 + s)) * 64 + d
                               : ((long)((bb * 16 + h) * 64 + d)) * 1024 + s;
        ((uint16_t*)out)[idx] = f32_to_bf16(v);
      }
    }
  }
}

// ---------- flash attention with relative-position bias ----------
// One wave per (b,h, 16-query-row tile).  Sk = 1024, hd = 64.
#define LDS_F_PER_WAVE (16 * 80 + 16 * 32)
__global__ __launch_bounds__(128)
void attn_kernel(const uint16_t* __restrict__ Q, const uint16_t* __restrict__ Km,
                 const uint16_t* __restrict__ Vt, const uint16_t* __restrict__ Eb,
                 uint16_t* __restrict__ O) {
  __shared__ float smemF[4 * LDS_F_PER_WAVE];
#if USE_ASYNC_LDS
  // per wave: K panel 2x(32 rows x 64 bf16) + V panel 2x(64 rows x 32 bf16) = 16KB
  __shared__ __attribute__((aligned(16))) uint16_t smemP[4 * 8192];
#endif
  int lane = threadIdx.x & 31;
  int wid  = threadIdx.x >> 5;
  int wave = (blockIdx.x * blockDim.x + threadIdx.x) >> 5;
  int bh = wave >> 6;                // 64 q-tiles per (b,h)
  int q0 = (wave & 63) * 16;
  int b = bh >> 4, h = bh & 15;
  int hi = lane >> 4, colLane = lane & 15;

  float* qE = smemF + wid * LDS_F_PER_WAVE;   // [16][80] q . E_r table
  float* Pb = qE + 16 * 80;                   // [16][32] softmax P bounce buffer

  const uint16_t* qBase = Q  + ((long)bh * 1024 + q0) * 64;
  const uint16_t* kBase = Km + (long)bh * 1024 * 64;
  const uint16_t* vBase = Vt + (long)bh * 64 * 1024;

  // q tile (16 x 64) as two A-fragments, resident for the whole wave
  const uint16_t* aRow = qBase + (long)colLane * 64;
  v16bf qa0 = load_frag_A(aRow + 0,  lane);
  v16bf qa1 = load_frag_A(aRow + 32, lane);

  // qE = q . E^T : 5 bucket tiles (80 padded buckets, 65 valid), 2 WMMAs each
#pragma unroll
  for (int t = 0; t < 5; ++t) {
    const uint16_t* ecol = Eb + (long)(t * 16 + colLane) * 64;
    v16bf b0 = load_frag_B(ecol + 0,  lane);
    v16bf b1 = load_frag_B(ecol + 32, lane);
    v8f c = {};
    c = WMMA_BF16(qa0, b0, c);
    c = WMMA_BF16(qa1, b1, c);
#pragma unroll
    for (int j = 0; j < 8; ++j)
      qE[(j + 8 * hi) * 80 + t * 16 + colLane] = c[j];
  }

#if USE_ASYNC_LDS
  uint16_t* kpan = smemP + wid * 8192;    // 2 x 2048
  uint16_t* vpan = kpan + 4096;           // 2 x 2048
  // 16 per-lane b128 async copies stage one (K,V) panel pair
  auto prefetch_panels = [&](int buf, int ks) {
    uint16_t* kp = kpan + buf * 2048;
    uint16_t* vp = vpan + buf * 2048;
#pragma unroll
    for (int i = 0; i < 8; ++i) {
      int c = lane + 32 * i;
      async_copy_b128(kBase + (long)(ks + (c >> 3)) * 64 + (c & 7) * 8, kp + c * 8);
      async_copy_b128(vBase + (long)(c >> 2) * 1024 + ks + (c & 3) * 8, vp + c * 8);
    }
  };
  prefetch_panels(0, 0);
#endif

  float mrow[8], lrow[8];
  v8f acc[4] = {};
#pragma unroll
  for (int j = 0; j < 8; ++j) { mrow[j] = -1e30f; lrow[j] = 0.f; }

  // Unrolled x2: LDS double-buffer parity becomes compile-time (immediate ds offsets).
#pragma unroll 2
  for (int ks = 0; ks < 1024; ks += 32) {
    int buf = (ks >> 5) & 1;
#if USE_ASYNC_LDS
    bool more = (ks + 32) < 1024;
    if (more) prefetch_panels(buf ^ 1, ks + 32);            // overlap next panel
    if (more) __builtin_amdgcn_s_wait_asynccnt(16);         // current panel landed
    else      __builtin_amdgcn_s_wait_asynccnt(0);
    const uint16_t* kP = kpan + buf * 2048;                 // row n -> kP + n*64
    const uint16_t* vP = vpan + buf * 2048;                 // row d -> vP + d*32
#else
    if (ks + 32 < 1024) {
      __builtin_prefetch(kBase + (long)(ks + 32) * 64, 0, 1);
      __builtin_prefetch(vBase + ks + 32, 0, 1);
    }
#endif
    // ---- gather all B-fragments for this step up front (distinct registers) ----
    v16bf kb0[2], kb1[2], vb[4];
#pragma unroll
    for (int kt = 0; kt < 2; ++kt) {
#if USE_ASYNC_LDS
      const uint16_t* kcol = kP + (long)(kt * 16 + colLane) * 64;
#else
      const uint16_t* kcol = kBase + (long)(ks + kt * 16 + colLane) * 64;
#endif
      kb0[kt] = load_frag_B(kcol + 0,  lane);
      kb1[kt] = load_frag_B(kcol + 32, lane);
    }
#pragma unroll
    for (int t = 0; t < 4; ++t) {
#if USE_ASYNC_LDS
      vb[t] = load_frag_B(vP + (long)(t * 16 + colLane) * 32, lane);
#else
      vb[t] = load_frag_B(vBase + (long)(t * 16 + colLane) * 1024 + ks, lane);
#endif
    }
    // ---- scores: two 16-col tiles, scale 1/sqrt(64), + gathered rel bias ----
    float sc[2][8];
#pragma unroll
    for (int kt = 0; kt < 2; ++kt) {
      v8f c = {};
      c = WMMA_BF16(qa0, kb0[kt], c);
      c = WMMA_BF16(qa1, kb1[kt], c);
#pragma unroll
      for (int j = 0; j < 8; ++j) {
        int r  = j + 8 * hi;
        int kq = (ks + kt * 16 + colLane) - (q0 + r);
        kq = (kq < -32) ? -32 : (kq > 32 ? 32 : kq);
        sc[kt][j] = c[j] * 0.125f + qE[r * 80 + (kq + 32)];
      }
    }
    // ---- online softmax (row reductions across the 16-lane half groups) ----
#pragma unroll
    for (int j = 0; j < 8; ++j) {
      float mx = fmaxf(sc[0][j], sc[1][j]);
#pragma unroll
      for (int d = 1; d < 16; d <<= 1) mx = fmaxf(mx, __shfl_xor(mx, d, 32));
      float mnew  = fmaxf(mrow[j], mx);
      float scale = __expf(mrow[j] - mnew);
      float p0 = __expf(sc[0][j] - mnew);
      float p1 = __expf(sc[1][j] - mnew);
      float ps = p0 + p1;
#pragma unroll
      for (int d = 1; d < 16; d <<= 1) ps += __shfl_xor(ps, d, 32);
      lrow[j] = lrow[j] * scale + ps;
      mrow[j] = mnew;
      int r = j + 8 * hi;
      Pb[r * 32 + colLane]      = p0;
      Pb[r * 32 + 16 + colLane] = p1;
#pragma unroll
      for (int t = 0; t < 4; ++t) acc[t][j] *= scale;
    }
    // ---- re-layout P (D-frag -> A-frag) through LDS, cast to bf16 ----
    v16bf pa;
    {
      int row = lane & 15;
      int c0  = hi * 8;
#pragma unroll
      for (int e = 0; e < 8; ++e) {
        pa[e]     = f32_to_bf16h(Pb[row * 32 + c0 + e]);
        pa[e + 8] = f32_to_bf16h(Pb[row * 32 + c0 + 16 + e]);
      }
    }
    // ---- acc += P @ V ----
#pragma unroll
    for (int t = 0; t < 4; ++t) acc[t] = WMMA_BF16(pa, vb[t], acc[t]);
  }
  // ---- epilogue: normalize and store O as bf16 [B,S,D] ----
#pragma unroll
  for (int j = 0; j < 8; ++j) {
    float inv = 1.f / lrow[j];
    int s = q0 + j + 8 * hi;
#pragma unroll
    for (int t = 0; t < 4; ++t) {
      int col = h * 64 + t * 16 + colLane;
      O[((long)(b * 1024 + s)) * 1024 + col] = f32_to_bf16(acc[t][j] * inv);
    }
  }
}

// ---------- host ----------
extern "C" void kernel_launch(void* const* d_in, const int* in_sizes, int n_in,
                              void* d_out, int out_size, void* d_ws, size_t ws_size,
                              hipStream_t stream) {
  const float* query = (const float*)d_in[0];
  const float* key   = (const float*)d_in[1];
  const float* value = (const float*)d_in[2];
  const float* Wq = (const float*)d_in[3];
  const float* bq = (const float*)d_in[4];
  const float* Wk = (const float*)d_in[5];
  const float* bk = (const float*)d_in[6];
  const float* Wv = (const float*)d_in[7];
  const float* bv = (const float*)d_in[8];
  const float* Wo = (const float*)d_in[9];
  const float* bo = (const float*)d_in[10];
  const float* rel = (const float*)d_in[11];

  const long BSD = 4l * 1024 * 1024;   // B*S*D elements
  const long DD  = 1024l * 1024;
  uint16_t* ws  = (uint16_t*)d_ws;
  uint16_t* xq  = ws;            // bf16 inputs; xq is reused for attention output O
  uint16_t* xk  = xq  + BSD;
  uint16_t* xv  = xk  + BSD;
  uint16_t* qb  = xv  + BSD;     // Q  [B,H,S,hd]
  uint16_t* kb  = qb  + BSD;     // K  [B,H,S,hd]
  uint16_t* vtb = kb  + BSD;     // V  [B,H,hd,S]
  uint16_t* wqb = vtb + BSD;
  uint16_t* wkb = wqb + DD;
  uint16_t* wvb = wkb + DD;
  uint16_t* wob = wvb + DD;
  uint16_t* eb  = wob + DD;      // rel enc padded to 80x64

  const int thr = 256;
  convert_f32_bf16<<<(int)((BSD + thr - 1) / thr), thr, 0, stream>>>(query, xq, (int)BSD);
  convert_f32_bf16<<<(int)((BSD + thr - 1) / thr), thr, 0, stream>>>(key,   xk, (int)BSD);
  convert_f32_bf16<<<(int)((BSD + thr - 1) / thr), thr, 0, stream>>>(value, xv, (int)BSD);
  convert_f32_bf16<<<(int)((DD  + thr - 1) / thr), thr, 0, stream>>>(Wq, wqb, (int)DD);
  convert_f32_bf16<<<(int)((DD  + thr - 1) / thr), thr, 0, stream>>>(Wk, wkb, (int)DD);
  convert_f32_bf16<<<(int)((DD  + thr - 1) / thr), thr, 0, stream>>>(Wv, wvb, (int)DD);
  convert_f32_bf16<<<(int)((DD  + thr - 1) / thr), thr, 0, stream>>>(Wo, wob, (int)DD);
  convert_pad_f32_bf16<<<(80 * 64 + thr - 1) / thr, thr, 0, stream>>>(rel, eb, 65 * 64, 80 * 64);

  dim3 blk(128);  // 4 wave32 per block
  // 4096 waves each: M=4096 rows in 16-row tiles x 16 column-strips of 64
  proj_gemm<<<1024, blk, 0, stream>>>(xq, wqb, bq, qb,  0);
  proj_gemm<<<1024, blk, 0, stream>>>(xk, wkb, bk, kb,  0);
  proj_gemm<<<1024, blk, 0, stream>>>(xv, wvb, bv, vtb, 1);

  // 4096 waves: B*H*(S/16) q-tiles
  attn_kernel<<<1024, blk, 0, stream>>>(qb, kb, vtb, eb, xq);

  proj_gemm<<<1024, blk, 0, stream>>>(xq, wob, bo, d_out, 2);
}
